// MixBlock_18949395710080
// MI455X (gfx1250) — matrix operand order repaired and checked by reading, not compiled, not run on hardware
//
#include <hip/hip_runtime.h>
#include <hip/hip_bf16.h>

// ---------------------------------------------------------------------------
// MixBlock for MI455X (gfx1250): dense GEMMs + attention on WMMA f16 (fp32
// accumulate); GEMM tiles staged by the Tensor Data Mover (double-buffered,
// TENSORcnt split-wait); attention K-chunks staged via async global->LDS
// (ASYNCcnt); softmax reductions on v_permlane16.
// ---------------------------------------------------------------------------

typedef __attribute__((ext_vector_type(16))) _Float16     v16h;
typedef __attribute__((ext_vector_type(8)))  _Float16     v8h;
typedef __attribute__((ext_vector_type(8)))  float        v8f;
typedef __attribute__((ext_vector_type(4)))  unsigned int v4u;
typedef __attribute__((ext_vector_type(8)))  unsigned int v8u;

#define DIMC  384
#define HEADS 8
#define HDIM  48
#define NB    8
#define NH    32
#define NW    32
#define NTOK  1024          // NH*NW
#define MTOT  8192          // NB*NTOK
#define HID   1536
#define QKVD  1152
#define BNEPS 1e-5f

union AFrag  { v16h h; unsigned int u[8]; };
union HVec   { v8h h; uint4 q; _Float16 e[8]; };

__device__ __forceinline__ v8f wmma16(const AFrag& a, const AFrag& b, v8f c) {
  // D = A(16x32 f16) x B(32x16 f16) + C(16x16 f32)
  return __builtin_amdgcn_wmma_f32_16x16x32_f16(false, a.h, false, b.h,
                                                (short)0, c, false, false);
}

// Generic->LDS raw offset: LDS aperture maps addr[31:0] to the LDS byte addr.
__device__ __forceinline__ unsigned lds_raw(const void* p) {
  return (unsigned)(unsigned long long)p;
}

// ---------------------------------------------------------------------------
// Tensor Data Mover: 2D f16 tile load, Global -> LDS (D# per ISA 8.3/8.4).
// ---------------------------------------------------------------------------
__device__ __forceinline__ void tdm_load_tile_f16(unsigned lds,
                                                  const _Float16* g,
                                                  unsigned tile_x,
                                                  unsigned tile_y,
                                                  unsigned stride_elems) {
  unsigned long long ga = (unsigned long long)g;
  v4u g0;
  g0[0] = 1u;                                   // count=1 (valid user D#)
  g0[1] = lds;                                  // lds_addr
  g0[2] = (unsigned)ga;                         // global_addr[31:0]
  g0[3] = (unsigned)((ga >> 32) & 0x01FFFFFFull) | (2u << 30);  // [56:32]|type=2
  const unsigned dim1 = 1u << 20;               // generous OOB bound
  v8u g1;
  g1[0] = 1u << 16;                             // data_size = 2 bytes
  g1[1] = (stride_elems & 0xFFFFu) << 16;       // tensor_dim0[15:0] @bit48
  g1[2] = (stride_elems >> 16) | ((dim1 & 0xFFFFu) << 16);
  g1[3] = (dim1 >> 16) | (tile_x << 16);        // tile_dim0 @bit112
  g1[4] = tile_y;                               // tile_dim1 (tile_dim2=0)
  g1[5] = stride_elems;                         // tensor_dim0_stride[31:0]
  g1[6] = 0u;
  g1[7] = 0u;
  asm volatile("tensor_load_to_lds %0, %1" ::"s"(g0), "s"(g1) : "memory");
}

__device__ __forceinline__ void tdm_wait0() {
#if __has_builtin(__builtin_amdgcn_s_wait_tensorcnt)
  __builtin_amdgcn_s_wait_tensorcnt(0);
#else
  asm volatile("s_wait_tensorcnt 0x0" ::: "memory");
#endif
}

// Async 16-byte global -> LDS copy (ASYNCcnt tracked, per-lane LDS address).
__device__ __forceinline__ void async_g2l_b128(unsigned lds,
                                               const void* gaddr) {
  asm volatile("global_load_async_to_lds_b128 %0, %1, off" ::"v"(lds),
               "v"((unsigned long long)gaddr)
               : "memory");
}

__device__ __forceinline__ void async_wait0() {
#if __has_builtin(__builtin_amdgcn_s_wait_asynccnt)
  __builtin_amdgcn_s_wait_asynccnt(0);
#else
  asm volatile("s_wait_asynccnt 0x0" ::: "memory");
#endif
}

// ---------------------------------------------------------------------------
// 16-lane butterfly reductions (rows of the WMMA C layout) on v_permlane16.
// ---------------------------------------------------------------------------
__device__ __forceinline__ float perm16(float x, unsigned s0, unsigned s1) {
#if __has_builtin(__builtin_amdgcn_permlane16)
  return __uint_as_float(__builtin_amdgcn_permlane16(
      __float_as_uint(x), __float_as_uint(x), s0, s1, false, false));
#else
  return __shfl_xor(x, 0, 32);  // unreachable fallback shape
#endif
}

__device__ __forceinline__ float red16_max(float x) {
#if __has_builtin(__builtin_amdgcn_permlane16)
  x = fmaxf(x, perm16(x, 0x67452301u, 0xEFCDAB89u));  // xor 1
  x = fmaxf(x, perm16(x, 0x54761032u, 0xDCFE98BAu));  // xor 2
  x = fmaxf(x, perm16(x, 0x32107654u, 0xBA98FEDCu));  // xor 4
  x = fmaxf(x, perm16(x, 0xFEDCBA98u, 0x76543210u));  // xor 8
#else
#pragma unroll
  for (int off = 1; off < 16; off <<= 1) x = fmaxf(x, __shfl_xor(x, off, 32));
#endif
  return x;
}

__device__ __forceinline__ float red16_sum(float x) {
#if __has_builtin(__builtin_amdgcn_permlane16)
  x += perm16(x, 0x67452301u, 0xEFCDAB89u);
  x += perm16(x, 0x54761032u, 0xDCFE98BAu);
  x += perm16(x, 0x32107654u, 0xBA98FEDCu);
  x += perm16(x, 0xFEDCBA98u, 0x76543210u);
#else
#pragma unroll
  for (int off = 1; off < 16; off <<= 1) x += __shfl_xor(x, off, 32);
#endif
  return x;
}

// ---------------------------------------------------------------------------
// Kernel 1: x1 = x + depthwise3x3(x); A16 = f16(BN1(x1)) token-major [m, c]
// ---------------------------------------------------------------------------
__global__ void pos_bn_kernel(const float* __restrict__ x,
                              const float* __restrict__ pw,
                              const float* __restrict__ pb,
                              const float* __restrict__ g1w,
                              const float* __restrict__ g1b,
                              const float* __restrict__ g1m,
                              const float* __restrict__ g1v,
                              float* __restrict__ x1,
                              _Float16* __restrict__ A16) {
  int idx = blockIdx.x * 256 + threadIdx.x;           // BCHW linear
  int w = idx & 31;
  int h = (idx >> 5) & 31;
  int c = (idx >> 10) % DIMC;
  int b = idx / (DIMC * NTOK);
  const float* xp = x + ((size_t)(b * DIMC + c) << 10);
  const float* wp = pw + c * 9;
  float acc = pb[c];
#pragma unroll
  for (int dy = -1; dy <= 1; ++dy) {
#pragma unroll
    for (int dx = -1; dx <= 1; ++dx) {
      int hh = h + dy, ww = w + dx;
      if (hh >= 0 && hh < NH && ww >= 0 && ww < NW)
        acc += wp[(dy + 1) * 3 + (dx + 1)] * xp[hh * NW + ww];
    }
  }
  float v = xp[h * NW + w] + acc;
  x1[idx] = v;
  float s  = g1w[c] * rsqrtf(g1v[c] + BNEPS);
  float sh = g1b[c] - g1m[c] * s;
  int m = b * NTOK + h * NW + w;
  A16[(size_t)m * DIMC + c] = (_Float16)(v * s + sh);
}

// ---------------------------------------------------------------------------
// Kernel 2: fold (optional) frozen-BN into a 1x1-conv weight, convert to f16.
// ---------------------------------------------------------------------------
__global__ void fold_kernel(const float* __restrict__ W,
                            const float* __restrict__ bias,
                            const float* __restrict__ bnw,
                            const float* __restrict__ bnb,
                            const float* __restrict__ bnm,
                            const float* __restrict__ bnv,
                            _Float16* __restrict__ W16,
                            float* __restrict__ beff, int K) {
  __shared__ float red[256];
  int n = blockIdx.x;
  int t = threadIdx.x;
  float part = 0.f;
  for (int k = t; k < K; k += 256) {
    float s = 1.f, sh = 0.f;
    if (bnw) {
      s  = bnw[k] * rsqrtf(bnv[k] + BNEPS);
      sh = bnb[k] - bnm[k] * s;
    }
    float wv = W[(size_t)n * K + k];
    W16[(size_t)n * K + k] = (_Float16)(wv * s);
    part += wv * sh;
  }
  red[t] = part;
  __syncthreads();
  for (int o = 128; o > 0; o >>= 1) {
    if (t < o) red[t] += red[t + o];
    __syncthreads();
  }
  if (t == 0) beff[n] = bias[n] + red[0];
}

// ---------------------------------------------------------------------------
// Kernel 3: WMMA GEMM with TDM-staged, double-buffered LDS tiles.
// ---------------------------------------------------------------------------
template <int MODE>
__global__ __launch_bounds__(256) void gemm_wmma(
    const _Float16* __restrict__ A, const _Float16* __restrict__ W,
    const float* __restrict__ bias, int K, int NN,
    float* __restrict__ out_f32, _Float16* __restrict__ out_f16,
    const float* __restrict__ aux) {
  __shared__ __align__(16) _Float16 As[2][128 * 32];
  __shared__ __align__(16) _Float16 Bs[2][128 * 32];

  int tid  = threadIdx.x;
  int wid  = tid >> 5;
  int lane = tid & 31;
  int ln   = lane & 15;
  int kgrp = lane >> 4;
  int waveM = wid & 1;
  int waveN = wid >> 1;
  int m0 = blockIdx.y * 128;
  int n0 = blockIdx.x * 128;

  v8f acc[4][2] = {};
  const int nk = K >> 5;

  if (wid == 0) {   // TDM prologue (TDM ignores EXEC; once per wave)
    tdm_load_tile_f16(lds_raw(&As[0][0]), A + (size_t)m0 * K, 32, 128, K);
    tdm_load_tile_f16(lds_raw(&Bs[0][0]), W + (size_t)n0 * K, 32, 128, K);
  }

#pragma unroll 2
  for (int kt = 0; kt < nk; ++kt) {
    int cur = kt & 1;
    if (wid == 0) tdm_wait0();        // current tile landed in LDS
    __syncthreads();                  // visible to all waves; prev reads done
    if (wid == 0 && kt + 1 < nk) {    // overlap next DMA with this compute
      int k0n = (kt + 1) << 5;
      tdm_load_tile_f16(lds_raw(&As[cur ^ 1][0]), A + (size_t)m0 * K + k0n,
                        32, 128, K);
      tdm_load_tile_f16(lds_raw(&Bs[cur ^ 1][0]), W + (size_t)n0 * K + k0n,
                        32, 128, K);
    }

    AFrag af[4], bf[2];
#pragma unroll
    for (int mt = 0; mt < 4; ++mt) {
      int row = waveM * 64 + mt * 16 + ln;
#pragma unroll
      for (int v = 0; v < 8; ++v) {
        int kb = ((v & 4) ? 16 : 0) + kgrp * 8 + (v & 3) * 2;  // A-layout
        af[mt].u[v] = *(const unsigned int*)&As[cur][row * 32 + kb];
      }
    }
#pragma unroll
    for (int nt = 0; nt < 2; ++nt) {
      int row = waveN * 32 + nt * 16 + ln;
#pragma unroll
      for (int v = 0; v < 8; ++v) {
        int kb = ((v & 4) ? 16 : 0) + kgrp * 8 + (v & 3) * 2;  // B-layout
        bf[nt].u[v] = *(const unsigned int*)&Bs[cur][row * 32 + kb];
      }
    }
#pragma unroll
    for (int mt = 0; mt < 4; ++mt)
#pragma unroll
      for (int nt = 0; nt < 2; ++nt)
        acc[mt][nt] = wmma16(af[mt], bf[nt], acc[mt][nt]);
  }

  // Epilogue: C layout — lane holds col = ln, rows = v + 8*kgrp
#pragma unroll
  for (int mt = 0; mt < 4; ++mt) {
#pragma unroll
    for (int nt = 0; nt < 2; ++nt) {
#pragma unroll
      for (int v = 0; v < 8; ++v) {
        int m = m0 + waveM * 64 + mt * 16 + v + kgrp * 8;
        int n = n0 + waveN * 32 + nt * 16 + ln;
        float cv = acc[mt][nt][v] + bias[n];
        if (MODE == 0) {
          out_f32[(size_t)m * NN + n] = cv;
          out_f16[(size_t)m * NN + n] = (_Float16)cv;
        } else if (MODE == 1) {
          int b = m >> 10, hw = m & 1023;
          size_t bi = ((size_t)(b * DIMC + n) << 10) + hw;
          float xv = aux[bi] + cv;
          out_f32[bi] = xv;
          out_f16[(size_t)m * NN + n] = (_Float16)xv;
        } else if (MODE == 2) {
          float g = 0.5f * cv * (1.f + erff(cv * 0.70710678118654752f));
          out_f16[(size_t)m * NN + n] = (_Float16)g;
        } else {  // MODE 3
          int b = m >> 10, hw = m & 1023;
          size_t bi = ((size_t)(b * DIMC + n) << 10) + hw;
          out_f32[bi] = aux[bi] + cv;
        }
      }
    }
  }
}

// ---------------------------------------------------------------------------
// Kernel 4: flash attention over f16 qkv copy. Block = (batch, head,
// 128-query tile), 8 waves, wave owns 16 query rows; key chunks of 32.
// K chunks staged with async global->LDS (ASYNCcnt); V transposed manually.
// ---------------------------------------------------------------------------
__global__ __launch_bounds__(256) void attn_kernel(
    const _Float16* __restrict__ qkv16, float* __restrict__ o) {
  __shared__ __align__(16) _Float16 Qs[128 * 64];   // queries, hd padded to 64
  __shared__ __align__(16) _Float16 Ks[32 * 64];    // key chunk [key][hd64]
  __shared__ __align__(16) _Float16 Vt[48 * 32];    // value chunk [hd][key]
  __shared__ __align__(16) _Float16 Ps[8][16 * 32]; // per-wave P transpose

  int b   = blockIdx.z;
  int hd  = blockIdx.y;
  int qt  = blockIdx.x;
  int tid = threadIdx.x;
  int wid = tid >> 5;
  int lane = tid & 31;
  int ln   = lane & 15;
  int kgrp = lane >> 4;
  const _Float16 hscale = (_Float16)0.144337567297406441f;  // 48^-0.5

  // stage Q (scaled, vectorized: 6 x 16B chunks per 48-half row)
  for (int u = tid; u < 128 * 6; u += 256) {
    int r = u / 6, seg = u % 6;
    HVec hv;
    hv.q = *(const uint4*)&qkv16[(size_t)(b * NTOK + qt * 128 + r) * QKVD +
                                 hd * HDIM + seg * 8];
    hv.h = hv.h * hscale;                      // v_pk_mul_f16 x4
    *(uint4*)&Qs[r * 64 + seg * 8] = hv.q;
  }
  // zero the pad columns (48..63) of Qs once; async K copies never touch
  // the Ks pad region either, so zero it once too.
  const uint4 z4 = {0u, 0u, 0u, 0u};
  for (int u = tid; u < 128 * 2; u += 256)
    *(uint4*)&Qs[(u >> 1) * 64 + 48 + (u & 1) * 8] = z4;
  if (tid < 64) *(uint4*)&Ks[(tid >> 1) * 64 + 48 + (tid & 1) * 8] = z4;
  __syncthreads();

  AFrag qa[2];
  int qrow = wid * 16 + ln;
#pragma unroll
  for (int kk = 0; kk < 2; ++kk)
#pragma unroll
    for (int v = 0; v < 8; ++v) {
      int kb = kk * 32 + ((v & 4) ? 16 : 0) + kgrp * 8 + (v & 3) * 2;
      qa[kk].u[v] = *(const unsigned int*)&Qs[qrow * 64 + kb];
    }

  v8f Oacc[3] = {};
  float mrun[8], lrun[8];
#pragma unroll
  for (int v = 0; v < 8; ++v) { mrun[v] = -3.0e38f; lrun[v] = 0.f; }

  for (int j0 = 0; j0 < NTOK; j0 += 32) {
    __syncthreads();                 // everyone done reading prev K/V
    // K chunk: 32 rows x 6 x 16B, async global->LDS, layout-preserving
    if (tid < 192) {
      int r = tid / 6, seg = tid % 6;
      async_g2l_b128(lds_raw(&Ks[r * 64 + seg * 8]),
                     &qkv16[(size_t)(b * NTOK + j0 + r) * QKVD + DIMC +
                            hd * HDIM + seg * 8]);
    }
    // V chunk: one uint4 per (key, 8-channel group), transposed into Vt
    if (tid < 192) {
      int r = tid / 6, seg = tid % 6;
      HVec hv;
      hv.q = *(const uint4*)&qkv16[(size_t)(b * NTOK + j0 + r) * QKVD +
                                   2 * DIMC + hd * HDIM + seg * 8];
#pragma unroll
      for (int i = 0; i < 8; ++i) Vt[(seg * 8 + i) * 32 + r] = hv.e[i];
    }
    async_wait0();                   // K landed in LDS
    __syncthreads();                 // all staging visible

    v8f S0 = {}, S1 = {};
#pragma unroll
    for (int kk = 0; kk < 2; ++kk) {
      AFrag kb0, kb1;
#pragma unroll
      for (int v = 0; v < 8; ++v) {
        int kb = kk * 32 + ((v & 4) ? 16 : 0) + kgrp * 8 + (v & 3) * 2;
        kb0.u[v] = *(const unsigned int*)&Ks[ln * 64 + kb];
        kb1.u[v] = *(const unsigned int*)&Ks[(ln + 16) * 64 + kb];
      }
      S0 = wmma16(qa[kk], kb0, S0);
      S1 = wmma16(qa[kk], kb1, S1);
    }

    // online softmax per row (row = v + 8*kgrp; 16 lanes share a row)
#pragma unroll
    for (int v = 0; v < 8; ++v) {
      float mx    = red16_max(fmaxf(S0[v], S1[v]));
      float mnew  = fmaxf(mrun[v], mx);
      float alpha = __expf(mrun[v] - mnew);
      mrun[v] = mnew;
      float p0 = __expf(S0[v] - mnew);
      float p1 = __expf(S1[v] - mnew);
      float ps = red16_sum(p0 + p1);
      lrun[v] = lrun[v] * alpha + ps;
#pragma unroll
      for (int nt = 0; nt < 3; ++nt) Oacc[nt][v] *= alpha;
      int prow = v + kgrp * 8;
      Ps[wid][prow * 32 + ln]      = (_Float16)p0;
      Ps[wid][prow * 32 + 16 + ln] = (_Float16)p1;
    }

    AFrag pa;
#pragma unroll
    for (int v = 0; v < 8; ++v) {
      int kb = ((v & 4) ? 16 : 0) + kgrp * 8 + (v & 3) * 2;
      pa.u[v] = *(const unsigned int*)&Ps[wid][ln * 32 + kb];
    }
#pragma unroll
    for (int nt = 0; nt < 3; ++nt) {
      AFrag vb;
#pragma unroll
      for (int v = 0; v < 8; ++v) {
        int kb = ((v & 4) ? 16 : 0) + kgrp * 8 + (v & 3) * 2;
        vb.u[v] = *(const unsigned int*)&Vt[(nt * 16 + ln) * 32 + kb];
      }
      Oacc[nt] = wmma16(pa, vb, Oacc[nt]);
    }
  }

#pragma unroll
  for (int nt = 0; nt < 3; ++nt)
#pragma unroll
    for (int v = 0; v < 8; ++v) {
      int row = qt * 128 + wid * 16 + v + kgrp * 8;
      float ov = Oacc[nt][v] / lrun[v];
      o[(size_t)(b * NTOK + row) * DIMC + hd * HDIM + nt * 16 + ln] = ov;
    }
}

// ---------------------------------------------------------------------------
// Kernel 5: depthwise 5x5 on v-slice of fp32 qkv + gated mix with attention.
// ---------------------------------------------------------------------------
__global__ void dwconv_mix_kernel(const float* __restrict__ qkv,
                                  const float* __restrict__ cw,
                                  const float* __restrict__ cb,
                                  const float* __restrict__ o,
                                  const float* __restrict__ saw,
                                  _Float16* __restrict__ mix16) {
  int idx = blockIdx.x * 256 + threadIdx.x;   // [m][c]
  int c = idx % DIMC;
  int m = idx / DIMC;
  int w = m & 31, h = (m >> 5) & 31, b = m >> 10;
  const float* wp = cw + c * 25;
  float acc = cb[c];
#pragma unroll
  for (int dy = -2; dy <= 2; ++dy) {
#pragma unroll
    for (int dx = -2; dx <= 2; ++dx) {
      int hh = h + dy, ww = w + dx;
      if (hh >= 0 && hh < NH && ww >= 0 && ww < NW) {
        int mm = b * NTOK + hh * NW + ww;
        acc += wp[(dy + 2) * 5 + (dx + 2)] *
               qkv[(size_t)mm * QKVD + 2 * DIMC + c];
      }
    }
  }
  float g = 1.f / (1.f + __expf(-saw[0]));
  mix16[idx] = (_Float16)(g * o[idx] + (1.f - g) * acc);
}

// ---------------------------------------------------------------------------
// Host-side orchestration
// ---------------------------------------------------------------------------
static inline size_t alignup(size_t v) { return (v + 255) & ~(size_t)255; }

extern "C" void kernel_launch(void* const* d_in, const int* in_sizes, int n_in,
                              void* d_out, int out_size, void* d_ws,
                              size_t ws_size, hipStream_t stream) {
  (void)in_sizes; (void)n_in; (void)out_size; (void)ws_size;
  const float* x       = (const float*)d_in[0];
  const float* pos_w   = (const float*)d_in[1];
  const float* pos_b   = (const float*)d_in[2];
  const float* bn1_w   = (const float*)d_in[3];
  const float* bn1_b   = (const float*)d_in[4];
  const float* bn1_m   = (const float*)d_in[5];
  const float* bn1_v   = (const float*)d_in[6];
  const float* conv1_w = (const float*)d_in[7];
  const float* conv1_b = (const float*)d_in[8];
  const float* conv_w  = (const float*)d_in[9];
  const float* conv_b  = (const float*)d_in[10];
  const float* conv2_w = (const float*)d_in[11];
  const float* conv2_b = (const float*)d_in[12];
  const float* bn2_w   = (const float*)d_in[13];
  const float* bn2_b   = (const float*)d_in[14];
  const float* bn2_m   = (const float*)d_in[15];
  const float* bn2_v   = (const float*)d_in[16];
  const float* fc1_w   = (const float*)d_in[17];
  const float* fc1_b   = (const float*)d_in[18];
  const float* fc2_w   = (const float*)d_in[19];
  const float* fc2_b   = (const float*)d_in[20];
  const float* sa_w    = (const float*)d_in[21];

  char* ws = (char*)d_ws;
  size_t off = 0;
  auto take = [&](size_t bytes) { size_t o = off; off = alignup(off + bytes); return o; };

  float*    X1    = (float*)   (ws + take((size_t)MTOT * DIMC * 4));
  _Float16* A16   = (_Float16*)(ws + take((size_t)MTOT * DIMC * 2));
  _Float16* W1    = (_Float16*)(ws + take((size_t)QKVD * DIMC * 2));
  float*    B1    = (float*)   (ws + take((size_t)QKVD * 4));
  _Float16* WC2   = (_Float16*)(ws + take((size_t)DIMC * DIMC * 2));
  float*    BC2   = (float*)   (ws + take((size_t)DIMC * 4));
  _Float16* W3    = (_Float16*)(ws + take((size_t)HID * DIMC * 2));
  float*    B3    = (float*)   (ws + take((size_t)HID * 4));
  _Float16* W4    = (_Float16*)(ws + take((size_t)DIMC * HID * 2));
  float*    B4    = (float*)   (ws + take((size_t)DIMC * 4));
  float*    QKV   = (float*)   (ws + take((size_t)MTOT * QKVD * 4));
  _Float16* QKV16 = (_Float16*)(ws + take((size_t)MTOT * QKVD * 2));
  float*    O     = (float*)   (ws + take((size_t)MTOT * DIMC * 4));
  _Float16* MIX   = (_Float16*)(ws + take((size_t)MTOT * DIMC * 2));
  float*    X2    = (float*)   (ws + take((size_t)MTOT * DIMC * 4));
  _Float16* A2    = (_Float16*)(ws + take((size_t)MTOT * DIMC * 2));
  _Float16* AG    = (_Float16*)(ws + take((size_t)MTOT * HID * 2));
  float*    OUT   = (float*)d_out;

  pos_bn_kernel<<<MTOT * DIMC / 256, 256, 0, stream>>>(
      x, pos_w, pos_b, bn1_w, bn1_b, bn1_m, bn1_v, X1, A16);

  fold_kernel<<<QKVD, 256, 0, stream>>>(conv1_w, conv1_b, bn1_w, bn1_b, bn1_m,
                                        bn1_v, W1, B1, DIMC);
  fold_kernel<<<DIMC, 256, 0, stream>>>(conv2_w, conv2_b, nullptr, nullptr,
                                        nullptr, nullptr, WC2, BC2, DIMC);
  fold_kernel<<<HID, 256, 0, stream>>>(fc1_w, fc1_b, bn2_w, bn2_b, bn2_m,
                                       bn2_v, W3, B3, DIMC);
  fold_kernel<<<DIMC, 256, 0, stream>>>(fc2_w, fc2_b, nullptr, nullptr,
                                        nullptr, nullptr, W4, B4, HID);

  gemm_wmma<0><<<dim3(QKVD / 128, MTOT / 128), 256, 0, stream>>>(
      A16, W1, B1, DIMC, QKVD, QKV, QKV16, nullptr);

  attn_kernel<<<dim3(NTOK / 128, HEADS, NB), 256, 0, stream>>>(QKV16, O);

  dwconv_mix_kernel<<<MTOT * DIMC / 256, 256, 0, stream>>>(QKV, conv_w, conv_b,
                                                           O, sa_w, MIX);

  gemm_wmma<1><<<dim3(DIMC / 128, MTOT / 128), 256, 0, stream>>>(
      MIX, WC2, BC2, DIMC, DIMC, X2, A2, X1);

  gemm_wmma<2><<<dim3(HID / 128, MTOT / 128), 256, 0, stream>>>(
      A2, W3, B3, DIMC, HID, nullptr, AG, nullptr);

  gemm_wmma<3><<<dim3(DIMC / 128, MTOT / 128), 256, 0, stream>>>(
      AG, W4, B4, HID, DIMC, OUT, nullptr, X2);
}
